// AdvancedSpectralGNN_25469156065943
// MI455X (gfx1250) — compile-verified, matrix operand order checked by reading, not run on hardware
//
#include <hip/hip_runtime.h>
#include <hip/hip_bf16.h>

// ---------------------------------------------------------------------------
// Types for CDNA5 WMMA
// ---------------------------------------------------------------------------
typedef __bf16 bf_t;
typedef __attribute__((ext_vector_type(16))) __bf16 v16bf;
typedef __attribute__((ext_vector_type(8)))  __bf16 v8bf;
typedef __attribute__((ext_vector_type(8)))  float  v8f;

#define HID 128
#define HEADS 4
#define HEAD_DIM 32
#define N_GRAPHS 1024
#define BN_EPS 1e-5f

// ---------------------------------------------------------------------------
// Helpers
// ---------------------------------------------------------------------------
__device__ __forceinline__ void atomAddF(float* p, float v) {
  __hip_atomic_fetch_add(p, v, __ATOMIC_RELAXED, __HIP_MEMORY_SCOPE_AGENT);
}

__device__ __forceinline__ bf_t f2bf(float f) {
  unsigned u = __builtin_bit_cast(unsigned, f);
  unsigned r = u + 0x7FFFu + ((u >> 16) & 1u);      // round-to-nearest-even
  unsigned short h = (unsigned short)(r >> 16);
  return __builtin_bit_cast(bf_t, h);
}

// monotone float <-> uint key for atomicMax-based segment max
__device__ __forceinline__ unsigned enc_key(float f) {
  unsigned b = __builtin_bit_cast(unsigned, f);
  return (b & 0x80000000u) ? ~b : (b | 0x80000000u);
}
__device__ __forceinline__ float dec_key(unsigned k) {
  unsigned b = (k & 0x80000000u) ? (k ^ 0x80000000u) : ~k;
  return __builtin_bit_cast(float, b);
}

__device__ __forceinline__ float elu1(float x) { return x > 0.f ? x : (__expf(x) - 1.f); }
__device__ __forceinline__ float lrelu02(float x) { return x < 0.f ? 0.2f * x : x; }

// ---------------------------------------------------------------------------
// f32 -> bf16 conversion
// ---------------------------------------------------------------------------
__global__ void to_bf16_kernel(const float* __restrict__ in, bf_t* __restrict__ out, int n) {
  int i = blockIdx.x * blockDim.x + threadIdx.x;
  if (i < n) out[i] = f2bf(in[i]);
}

// ---------------------------------------------------------------------------
// C[N x 128] = A_bf16[N x 128] @ W_f32[128 x 128]   (bf16 WMMA, f32 accum)
// One wave = 16 rows x 128 cols (8 accumulator tiles), K loop of 4 x 32.
// ---------------------------------------------------------------------------
__global__ __launch_bounds__(256)
void gemm_n128_kernel(const bf_t* __restrict__ A, const float* __restrict__ W,
                      float* __restrict__ C, int nRows) {
  __shared__ bf_t Wt[128 * 136];           // W^T, [n][k], padded rows (bank spread)
  for (int i = threadIdx.x; i < 128 * 128; i += 256) {
    int k = i >> 7, n = i & 127;           // W row-major [k][n]
    Wt[n * 136 + k] = f2bf(W[i]);
  }
  __syncthreads();

  const int wave = threadIdx.x >> 5;
  const int lane = threadIdx.x & 31;
  const int rowBase = (blockIdx.x * 8 + wave) * 16;
  if (rowBase >= nRows) return;            // wave-uniform: EXEC stays all-ones

  const int m16 = lane & 15;
  const int hi  = lane >> 4;               // half-wave selects K sub-chunk

  v8f acc[8] = {};
#pragma unroll
  for (int kt = 0; kt < 4; ++kt) {
    const int ka = kt * 32 + hi * 8;       // lane's first K chunk; second at +16
    const bf_t* ap = A + (size_t)(rowBase + m16) * 128 + ka;
    v8bf a0 = *(const v8bf*)ap;
    v8bf a1 = *(const v8bf*)(ap + 16);
    v16bf afrag;
#pragma unroll
    for (int i = 0; i < 8; ++i) { afrag[i] = a0[i]; afrag[8 + i] = a1[i]; }
#pragma unroll
    for (int nt = 0; nt < 8; ++nt) {
      const bf_t* bp = &Wt[(nt * 16 + m16) * 136 + ka];
      v8bf b0 = *(const v8bf*)bp;
      v8bf b1 = *(const v8bf*)(bp + 16);
      v16bf bfrag;
#pragma unroll
      for (int i = 0; i < 8; ++i) { bfrag[i] = b0[i]; bfrag[8 + i] = b1[i]; }
      acc[nt] = __builtin_amdgcn_wmma_f32_16x16x32_bf16(
          false, afrag, false, bfrag, (short)0, acc[nt], false, false);
    }
  }
  // D layout: VGPR r -> row rowBase + r + (hi?8:0), col nt*16 + m16
  const int srow = rowBase + hi * 8;
#pragma unroll
  for (int nt = 0; nt < 8; ++nt) {
    float* cp = C + (size_t)srow * 128 + nt * 16 + m16;
#pragma unroll
    for (int r = 0; r < 8; ++r) cp[(size_t)r * 128] = acc[nt][r];
  }
}

// ---------------------------------------------------------------------------
// GAT attention coefficients: a_src[n,h], a_dst[n,h]
// ---------------------------------------------------------------------------
__global__ void attn_coef_kernel(const float* __restrict__ h,
                                 const float* __restrict__ att_src,
                                 const float* __restrict__ att_dst,
                                 float* __restrict__ asrc, float* __restrict__ adst, int n) {
  int t = blockIdx.x * blockDim.x + threadIdx.x;
  if (t >= n * HEADS) return;
  int node = t >> 2, hd = t & 3;
  const float* hp = h + (size_t)node * HID + hd * HEAD_DIM;
  float s1 = 0.f, s2 = 0.f;
#pragma unroll 8
  for (int d = 0; d < HEAD_DIM; ++d) {
    float v = hp[d];
    s1 += v * att_src[hd * HEAD_DIM + d];
    s2 += v * att_dst[hd * HEAD_DIM + d];
  }
  asrc[t] = s1;
  adst[t] = s2;
}

// ---------------------------------------------------------------------------
// Degree (dst, incl. self loops) / inverse sqrt
// ---------------------------------------------------------------------------
__global__ void deg_kernel(const long long* __restrict__ ei, float* __restrict__ deg,
                           int E, int N) {
  int e = blockIdx.x * blockDim.x + threadIdx.x;
  if (e >= E + N) return;
  int dst = (e < E) ? (int)ei[E + e] : (e - E);
  atomAddF(&deg[dst], 1.f);
}
__global__ void dinv_kernel(float* __restrict__ deg, int N) {
  int i = blockIdx.x * blockDim.x + threadIdx.x;
  if (i >= N) return;
  float d = deg[i];
  deg[i] = d > 0.f ? __frsqrt_rn(d) : 0.f;
}

// ---------------------------------------------------------------------------
// GAT softmax passes over edges(+self loops), per (edge, head)
// ---------------------------------------------------------------------------
__global__ void gat_max_kernel(const long long* __restrict__ ei,
                               const float* __restrict__ asrc, const float* __restrict__ adst,
                               unsigned* __restrict__ m_u, int E, int N) {
  int t = blockIdx.x * blockDim.x + threadIdx.x;
  if (t >= (E + N) * HEADS) return;
  int e = t >> 2, hd = t & 3;
  int src = (e < E) ? (int)ei[e] : (e - E);
  int dst = (e < E) ? (int)ei[E + e] : (e - E);
  float x = lrelu02(asrc[src * HEADS + hd] + adst[dst * HEADS + hd]);
  atomicMax(&m_u[dst * HEADS + hd], enc_key(x));
}

__global__ void gat_sum_kernel(const long long* __restrict__ ei,
                               const float* __restrict__ asrc, const float* __restrict__ adst,
                               const unsigned* __restrict__ m_u, float* __restrict__ denom,
                               int E, int N) {
  int t = blockIdx.x * blockDim.x + threadIdx.x;
  if (t >= (E + N) * HEADS) return;
  int e = t >> 2, hd = t & 3;
  int src = (e < E) ? (int)ei[e] : (e - E);
  int dst = (e < E) ? (int)ei[E + e] : (e - E);
  float x = lrelu02(asrc[src * HEADS + hd] + adst[dst * HEADS + hd]);
  float m = dec_key(m_u[dst * HEADS + hd]);
  atomAddF(&denom[dst * HEADS + hd], __expf(x - m));
}

// one wave per edge: 32 lanes x float4 = 128 features, per-head alpha
__global__ __launch_bounds__(256)
void gat_scatter_kernel(const long long* __restrict__ ei, const float* __restrict__ h,
                        const float* __restrict__ asrc, const float* __restrict__ adst,
                        const unsigned* __restrict__ m_u, const float* __restrict__ denom,
                        float* __restrict__ out, int E, int N) {
  int wid = (int)((blockIdx.x * (size_t)blockDim.x + threadIdx.x) >> 5);
  int lane = threadIdx.x & 31;
  if (wid >= E + N) return;
  int src = (wid < E) ? (int)ei[wid] : (wid - E);
  int dst = (wid < E) ? (int)ei[E + wid] : (wid - E);
  int hd = lane >> 3;                       // 8 lanes per head
  float x = lrelu02(asrc[src * HEADS + hd] + adst[dst * HEADS + hd]);
  float m = dec_key(m_u[dst * HEADS + hd]);
  float alpha = __expf(x - m) / denom[dst * HEADS + hd];
  const float4 v = *(const float4*)(h + (size_t)src * HID + lane * 4);
  float* o = out + (size_t)dst * HID + lane * 4;
  atomAddF(o + 0, v.x * alpha);
  atomAddF(o + 1, v.y * alpha);
  atomAddF(o + 2, v.z * alpha);
  atomAddF(o + 3, v.w * alpha);
}

// ---------------------------------------------------------------------------
// GCN scatter: out[dst] += dinv[src]*dinv[dst] * xw[src]; one wave per edge
// ---------------------------------------------------------------------------
__global__ __launch_bounds__(256)
void gcn_scatter_kernel(const long long* __restrict__ ei, const float* __restrict__ xw,
                        const float* __restrict__ dinv, float* __restrict__ out,
                        int E, int N) {
  int wid = (int)((blockIdx.x * (size_t)blockDim.x + threadIdx.x) >> 5);
  int lane = threadIdx.x & 31;
  if (wid >= E + N) return;
  int src = (wid < E) ? (int)ei[wid] : (wid - E);
  int dst = (wid < E) ? (int)ei[E + wid] : (wid - E);
  float nrm = dinv[src] * dinv[dst];
  const float4 v = *(const float4*)(xw + (size_t)src * HID + lane * 4);
  float* o = out + (size_t)dst * HID + lane * 4;
  atomAddF(o + 0, v.x * nrm);
  atomAddF(o + 1, v.y * nrm);
  atomAddF(o + 2, v.z * nrm);
  atomAddF(o + 3, v.w * nrm);
}

// ---------------------------------------------------------------------------
// GAT finalize: h = elu(scatter + gat_b)      (in place on hB)
// ---------------------------------------------------------------------------
__global__ void gat_post_kernel(float* __restrict__ hB, const float* __restrict__ bias, int n128) {
  int i = blockIdx.x * blockDim.x + threadIdx.x;
  if (i >= n128) return;
  hB[i] = elu1(hB[i] + bias[i & 127]);
}

// GCN post: hB = elu(bn(scat + b)) (+ hB if residual)
__global__ void gcn_post_kernel(float* __restrict__ hB, const float* __restrict__ scat,
                                const float* __restrict__ gb,
                                const float* __restrict__ bg, const float* __restrict__ bb,
                                const float* __restrict__ bm, const float* __restrict__ bv,
                                int n128, int addResid) {
  int i = blockIdx.x * blockDim.x + threadIdx.x;
  if (i >= n128) return;
  int c = i & 127;
  float t = scat[i] + gb[c];
  t = (t - bm[c]) * __frsqrt_rn(bv[c] + BN_EPS) * bg[c] + bb[c];
  t = elu1(t);
  if (addResid) t += hB[i];
  hB[i] = t;
}

// ---------------------------------------------------------------------------
// Mean pool scatter: one wave per node
// ---------------------------------------------------------------------------
__global__ __launch_bounds__(256)
void pool_kernel(const float* __restrict__ h, const long long* __restrict__ batch,
                 float* __restrict__ sums, float* __restrict__ cnt, int N) {
  int node = (int)((blockIdx.x * (size_t)blockDim.x + threadIdx.x) >> 5);
  int lane = threadIdx.x & 31;
  if (node >= N) return;
  int g = (int)batch[node];
  const float4 v = *(const float4*)(h + (size_t)node * HID + lane * 4);
  float* o = sums + (size_t)g * HID + lane * 4;
  atomAddF(o + 0, v.x);
  atomAddF(o + 1, v.y);
  atomAddF(o + 2, v.z);
  atomAddF(o + 3, v.w);
  if (lane == 0) atomAddF(&cnt[g], 1.f);
}

// ---------------------------------------------------------------------------
// MLP head: per graph (block of 128 threads)
// ---------------------------------------------------------------------------
__global__ __launch_bounds__(128)
void mlp_kernel(const float* __restrict__ sums, const float* __restrict__ cnt,
                const float* __restrict__ p1w, const float* __restrict__ p1b,
                const float* __restrict__ p2w, const float* __restrict__ p2b,
                const float* __restrict__ cw,  const float* __restrict__ cb,
                float* __restrict__ outp, float* __restrict__ embp) {
  __shared__ float g[128], e1[128], emb[64];
  int b = blockIdx.x, t = threadIdx.x;
  float c = cnt[b]; c = c > 1.f ? c : 1.f;
  g[t] = sums[(size_t)b * 128 + t] / c;
  __syncthreads();
  float s = p1b[t];
  for (int k = 0; k < 128; ++k) s += g[k] * p1w[k * 128 + t];
  e1[t] = elu1(s);
  __syncthreads();
  if (t < 64) {
    float s2 = p2b[t];
    for (int k = 0; k < 128; ++k) s2 += e1[k] * p2w[k * 64 + t];
    emb[t] = s2;
    embp[(size_t)b * 64 + t] = s2;
  }
  __syncthreads();
  if (t < 3) {
    float s3 = cb[t];
    for (int k = 0; k < 64; ++k) s3 += emb[k] * cw[k * 3 + t];
    outp[(size_t)b * 3 + t] = s3;
  }
}

// ---------------------------------------------------------------------------
// Launch
// ---------------------------------------------------------------------------
extern "C" void kernel_launch(void* const* d_in, const int* in_sizes, int n_in,
                              void* d_out, int out_size, void* d_ws, size_t ws_size,
                              hipStream_t stream) {
  const int N = in_sizes[0] / HID;        // 100000
  const int E = in_sizes[1] / 2;          // 1600000
  const size_t nf = (size_t)N * HID;

  const float*     x     = (const float*)d_in[0];
  const long long* ei    = (const long long*)d_in[1];
  const long long* batch = (const long long*)d_in[3];
  const float* gat_w   = (const float*)d_in[4];
  const float* att_src = (const float*)d_in[5];
  const float* att_dst = (const float*)d_in[6];
  const float* gat_b   = (const float*)d_in[7];
  const float* gcn_w[3] = {(const float*)d_in[8],  (const float*)d_in[14], (const float*)d_in[20]};
  const float* gcn_b[3] = {(const float*)d_in[9],  (const float*)d_in[15], (const float*)d_in[21]};
  const float* bn_g[3]  = {(const float*)d_in[10], (const float*)d_in[16], (const float*)d_in[22]};
  const float* bn_b[3]  = {(const float*)d_in[11], (const float*)d_in[17], (const float*)d_in[23]};
  const float* bn_m[3]  = {(const float*)d_in[12], (const float*)d_in[18], (const float*)d_in[24]};
  const float* bn_v[3]  = {(const float*)d_in[13], (const float*)d_in[19], (const float*)d_in[25]};
  const float* p1w = (const float*)d_in[26];
  const float* p1b = (const float*)d_in[27];
  const float* p2w = (const float*)d_in[28];
  const float* p2b = (const float*)d_in[29];
  const float* cw  = (const float*)d_in[30];
  const float* cb  = (const float*)d_in[31];

  // workspace layout
  char* ws = (char*)d_ws;
  size_t need = 0;
  auto alloc = [&](size_t bytes) {
    char* p = ws + need;
    need += (bytes + 255) & ~(size_t)255;
    return (void*)p;
  };
  float*    bufA  = (float*)alloc(nf * 4);              // h / xw
  float*    bufB  = (float*)alloc(nf * 4);              // current node features
  float*    bufC  = (float*)alloc(nf * 4);              // scatter accumulator
  bf_t*     bufBF = (bf_t*)alloc(nf * 2);               // bf16 matmul input
  float*    asrc  = (float*)alloc((size_t)N * HEADS * 4);
  float*    adst  = (float*)alloc((size_t)N * HEADS * 4);
  unsigned* m_u   = (unsigned*)alloc((size_t)N * HEADS * 4);
  float*    denom = (float*)alloc((size_t)N * HEADS * 4);
  float*    dinv  = (float*)alloc((size_t)N * 4);
  float*    sums  = (float*)alloc((size_t)(N_GRAPHS * HID + N_GRAPHS) * 4); // sums||cnt
  float*    cnt   = sums + (size_t)N_GRAPHS * HID;
  if (need > ws_size) return;  // insufficient scratch: bail deterministically

  float* outp = (float*)d_out;
  float* embp = outp + (size_t)N_GRAPHS * 3;

  const int TPB = 256;
  const int gridNf   = (int)((nf + TPB - 1) / TPB);
  const int gridGemm = (N + 127) / 128;
  const int EN = E + N;
  const int gridEN4  = (int)(((size_t)EN * HEADS + TPB - 1) / TPB);
  const int gridENw  = (int)(((size_t)EN * 32 + TPB - 1) / TPB);
  const int gridNw   = (int)(((size_t)N * 32 + TPB - 1) / TPB);

  // zero scatter targets / reductions
  hipMemsetAsync(bufB,  0, nf * 4, stream);
  hipMemsetAsync(m_u,   0, (size_t)N * HEADS * 4, stream);     // key 0 == -inf
  hipMemsetAsync(denom, 0, (size_t)N * HEADS * 4, stream);
  hipMemsetAsync(dinv,  0, (size_t)N * 4, stream);
  hipMemsetAsync(sums,  0, (size_t)(N_GRAPHS * HID + N_GRAPHS) * 4, stream);

  // ---- GAT: h = x @ gat_w (bf16 WMMA), attention softmax, scatter ----
  to_bf16_kernel<<<gridNf, TPB, 0, stream>>>(x, bufBF, (int)nf);
  gemm_n128_kernel<<<gridGemm, TPB, 0, stream>>>(bufBF, gat_w, bufA, N);
  attn_coef_kernel<<<(N * HEADS + TPB - 1) / TPB, TPB, 0, stream>>>(bufA, att_src, att_dst, asrc, adst, N);
  deg_kernel<<<(EN + TPB - 1) / TPB, TPB, 0, stream>>>(ei, dinv, E, N);
  dinv_kernel<<<(N + TPB - 1) / TPB, TPB, 0, stream>>>(dinv, N);
  gat_max_kernel<<<gridEN4, TPB, 0, stream>>>(ei, asrc, adst, m_u, E, N);
  gat_sum_kernel<<<gridEN4, TPB, 0, stream>>>(ei, asrc, adst, m_u, denom, E, N);
  gat_scatter_kernel<<<gridENw, TPB, 0, stream>>>(ei, bufA, asrc, adst, m_u, denom, bufB, E, N);
  gat_post_kernel<<<gridNf, TPB, 0, stream>>>(bufB, gat_b, (int)nf);

  // ---- 3 GCN layers ----
  for (int l = 0; l < 3; ++l) {
    to_bf16_kernel<<<gridNf, TPB, 0, stream>>>(bufB, bufBF, (int)nf);
    gemm_n128_kernel<<<gridGemm, TPB, 0, stream>>>(bufBF, gcn_w[l], bufA, N);
    hipMemsetAsync(bufC, 0, nf * 4, stream);
    gcn_scatter_kernel<<<gridENw, TPB, 0, stream>>>(ei, bufA, dinv, bufC, E, N);
    gcn_post_kernel<<<gridNf, TPB, 0, stream>>>(bufB, bufC, gcn_b[l],
                                                bn_g[l], bn_b[l], bn_m[l], bn_v[l],
                                                (int)nf, l < 2 ? 1 : 0);
  }

  // ---- pool + MLP head ----
  pool_kernel<<<gridNw, TPB, 0, stream>>>(bufB, batch, sums, cnt, N);
  mlp_kernel<<<N_GRAPHS, 128, 0, stream>>>(sums, cnt, p1w, p1b, p2w, p2b, cw, cb, outp, embp);
}
